// SQAE_10574209483429
// MI455X (gfx1250) — matrix operand order repaired and checked by reading, not compiled, not run on hardware
//
#include <hip/hip_runtime.h>
#include <hip/hip_bf16.h>
#include <stdint.h>

typedef __attribute__((ext_vector_type(2)))  float  v2f;
typedef __attribute__((ext_vector_type(4)))  float  v4f;
typedef __attribute__((ext_vector_type(8)))  float  v8f;
typedef __attribute__((ext_vector_type(16))) __bf16 v16bf;

#define INPUT_DIM 1024
#define LATENT    256
#define NUMEMB    8192
#define NTOT      32768

__device__ __forceinline__ v8f zero8() {
  v8f r;
#pragma unroll
  for (int i = 0; i < 8; ++i) r[i] = 0.f;
  return r;
}

// Round-to-nearest bf16 split: x ~= hi + lo with ~16 effective mantissa bits.
__device__ __forceinline__ void split_bf16(float x, unsigned short& hi, unsigned short& lo) {
  unsigned u  = __float_as_uint(x);
  unsigned r  = u + 0x7FFFu + ((u >> 16) & 1u);
  unsigned short h = (unsigned short)(r >> 16);
  float hf = __uint_as_float(((unsigned)h) << 16);
  float lf = x - hf;
  unsigned ul = __float_as_uint(lf);
  unsigned rl = ul + 0x7FFFu + ((ul >> 16) & 1u);
  hi = h;
  lo = (unsigned short)(rl >> 16);
}

// CDNA5 async Global->LDS (ASYNCcnt-tracked), two b128s per call: bytes
// [0,16) and [4096,4112) relative to both addresses.
__device__ __forceinline__ void async_copy_pair(unsigned lds_off, const void* gaddr) {
  asm volatile(
      "global_load_async_to_lds_b128 %0, %1, off\n\t"
      "global_load_async_to_lds_b128 %0, %1, off offset:4096"
      :: "v"(lds_off), "v"(gaddr) : "memory");
}

__device__ __forceinline__ void wait_async0() {
  asm volatile("s_wait_asynccnt 0x0" ::: "memory");
}

// ---------------------------------------------------------------------------
// Kernel 0: pack codebook E into WMMA-B-ready split-bf16 layout + 0.5*||E_k||^2
// B order chosen here for a 32x16 tile (contiguous k, same as A): lane l -> column n=l&15, k-base (l>>4)*16,
// 16 halves per lane are consecutive k. Chunk (ctile,q) is 512 halves.
// ---------------------------------------------------------------------------
__global__ void vq_prep(const float* __restrict__ E,
                        unsigned short* __restrict__ ehi,
                        unsigned short* __restrict__ elo,
                        float* __restrict__ ehn) {
  int c = blockIdx.x * blockDim.x + threadIdx.x;
  if (c >= NUMEMB) return;
  const float* row = E + (size_t)c * LATENT;
  int ct = c >> 4, n = c & 15;
  float s = 0.f;
  for (int k = 0; k < LATENT; ++k) {
    float e = row[k];
    s += e * e;
    unsigned short h, l16;
    split_bf16(e, h, l16);
    int q = k >> 5, r5 = k & 31;
    int hb = r5 >> 4, hh = r5 & 15;
    int lane = hb * 16 + n;
    size_t p = (size_t)(ct * 8 + q) * 512 + (size_t)lane * 16 + hh;
    ehi[p] = h;
    elo[p] = l16;
  }
  ehn[c] = 0.5f * s;
}

// ---------------------------------------------------------------------------
// Kernel 1: encoder z = x @ We^T + be, exact f32 via V_WMMA_F32_16X16X4_F32.
// One wave computes a full 16x256 strip (16 C tiles, 128 acc VGPRs); x is
// streamed exactly once, We (1 MB) stays hot in cache.
// ---------------------------------------------------------------------------
__global__ void __launch_bounds__(128) vq_encode(const float* __restrict__ x,
                                                 const float* __restrict__ We,
                                                 const float* __restrict__ be,
                                                 float* __restrict__ z) {
  int lane = threadIdx.x & 31, wave = threadIdx.x >> 5;
  int m0 = (blockIdx.x * 4 + wave) * 16;
  int mrow = m0 + (lane & 15);
  int koff = 2 * (lane >> 4);
  v8f acc[16];
#pragma unroll
  for (int t = 0; t < 16; ++t) acc[t] = zero8();

  for (int k0 = 0; k0 < INPUT_DIM; k0 += 4) {
    v2f a = *(const v2f*)(x + (size_t)mrow * INPUT_DIM + k0 + koff);
#pragma unroll
    for (int t = 0; t < 16; ++t) {
      int n = t * 16 + (lane & 15);
      v2f b = *(const v2f*)(We + (size_t)n * INPUT_DIM + k0 + koff);
      acc[t] = __builtin_amdgcn_wmma_f32_16x16x4_f32(false, a, false, b,
                                                     (short)0, acc[t], false, false);
    }
  }
#pragma unroll
  for (int t = 0; t < 16; ++t) {
    int n = t * 16 + (lane & 15);
    float bias = be[n];
#pragma unroll
    for (int j = 0; j < 8; ++j) {
      int row = m0 + j + ((lane >> 4) << 3);
      z[(size_t)row * LATENT + n] = acc[t][j] + bias;
    }
  }
}

// ---------------------------------------------------------------------------
// Kernel 2: nearest-codeword search. score = 0.5*||E_k||^2 - z.E_k (||z||^2 is
// row-constant, dropped). z rows held in registers as split-bf16 A fragments
// (8 chunks x (hi,lo)); E tiles double-buffered in LDS via CDNA5 async
// global->LDS loads; 3 bf16 WMMAs per K-chunk. Then cross-lane argmin
// reduction and z_q gather.
// ---------------------------------------------------------------------------
__global__ void __launch_bounds__(256) vq_assign(const float* __restrict__ z,
                                                 const unsigned short* __restrict__ ehi,
                                                 const unsigned short* __restrict__ elo,
                                                 const float* __restrict__ ehn,
                                                 const float* __restrict__ E,
                                                 int* __restrict__ idx_out,
                                                 float* __restrict__ zq) {
  __shared__ unsigned short sHi[2][4096];
  __shared__ unsigned short sLo[2][4096];
  __shared__ int sIdx[8][16];

  int lane = threadIdx.x & 31, wave = threadIdx.x >> 5;
  int m0 = (blockIdx.x * 8 + wave) * 16;
  int mrow = m0 + (lane & 15);
  int hb = lane >> 4;

  // Build A fragments: halves 0..15 = K hb*16+0..15 within each 32-wide K
  // chunk (matches packed B layout).
  v16bf Ahi[8], Alo[8];
#pragma unroll
  for (int q = 0; q < 8; ++q) {
    union { v16bf v; unsigned short u[16]; } ch, cl;
#pragma unroll
    for (int g = 0; g < 2; ++g) {
      int kbase = q * 32 + hb * 16 + g * 8;
      const float* src = z + (size_t)mrow * LATENT + kbase;
#pragma unroll
      for (int e = 0; e < 8; ++e) {
        unsigned short h, l16;
        split_bf16(src[e], h, l16);
        ch.u[g * 8 + e] = h;
        cl.u[g * 8 + e] = l16;
      }
    }
    Ahi[q] = ch.v;
    Alo[q] = cl.v;
  }

  float best[8];
  int bidx[8];
#pragma unroll
  for (int j = 0; j < 8; ++j) { best[j] = 3.4e38f; bidx[j] = 0; }

  // LDS byte offsets for this thread's async-copy slots in each buffer.
  unsigned hOff[2], lOff[2];
  hOff[0] = (unsigned)(uintptr_t)&sHi[0][0] + threadIdx.x * 16u;
  hOff[1] = (unsigned)(uintptr_t)&sHi[1][0] + threadIdx.x * 16u;
  lOff[0] = (unsigned)(uintptr_t)&sLo[0][0] + threadIdx.x * 16u;
  lOff[1] = (unsigned)(uintptr_t)&sLo[1][0] + threadIdx.x * 16u;

  // Prologue: async-stage tile 0 into buffer 0.
  {
    const char* gh = (const char*)(ehi) + threadIdx.x * 16u;
    const char* gl = (const char*)(elo) + threadIdx.x * 16u;
    async_copy_pair(hOff[0], gh);
    async_copy_pair(lOff[0], gl);
  }
  wait_async0();
  __syncthreads();

  for (int ct = 0; ct < 512; ++ct) {
    int buf = ct & 1;
    if (ct + 1 < 512) {
      const char* gh = (const char*)(ehi + (size_t)(ct + 1) * 4096) + threadIdx.x * 16u;
      const char* gl = (const char*)(elo + (size_t)(ct + 1) * 4096) + threadIdx.x * 16u;
      async_copy_pair(hOff[buf ^ 1], gh);
      async_copy_pair(lOff[buf ^ 1], gl);
    }

    v8f acc = zero8();
#pragma unroll
    for (int q = 0; q < 8; ++q) {
      v16bf bh = *(const v16bf*)(&sHi[buf][q * 512 + lane * 16]);
      v16bf bl = *(const v16bf*)(&sLo[buf][q * 512 + lane * 16]);
      acc = __builtin_amdgcn_wmma_f32_16x16x32_bf16(false, Ahi[q], false, bh, (short)0, acc, false, false);
      acc = __builtin_amdgcn_wmma_f32_16x16x32_bf16(false, Alo[q], false, bh, (short)0, acc, false, false);
      acc = __builtin_amdgcn_wmma_f32_16x16x32_bf16(false, Ahi[q], false, bl, (short)0, acc, false, false);
    }

    int n = ct * 16 + (lane & 15);
    float eh = ehn[n];
#pragma unroll
    for (int j = 0; j < 8; ++j) {
      float sc = eh - acc[j];
      if (sc < best[j]) { best[j] = sc; bidx[j] = n; }  // ascending n: strict < keeps lowest idx
    }

    wait_async0();     // this wave's loads into buf^1 are in LDS
    __syncthreads();   // everyone's loads are in LDS
  }

  // Row j lives on lanes 0..15 (VGPR j) and row 8+j on lanes 16..31; reduce
  // across the 16 column-lanes with xor shuffles (masks stay within a half).
#pragma unroll
  for (int j = 0; j < 8; ++j) {
#pragma unroll
    for (int off = 1; off < 16; off <<= 1) {
      float ob = __shfl_xor(best[j], off, 32);
      int   oi = __shfl_xor(bidx[j], off, 32);
      if (ob < best[j] || (ob == best[j] && oi < bidx[j])) { best[j] = ob; bidx[j] = oi; }
    }
  }
  if (lane == 0)  { for (int j = 0; j < 8; ++j) sIdx[wave][j]     = bidx[j]; }
  if (lane == 16) { for (int j = 0; j < 8; ++j) sIdx[wave][8 + j] = bidx[j]; }
  __syncthreads();
  if (lane == 0) {
    for (int j = 0; j < 16; ++j) idx_out[m0 + j] = sIdx[wave][j];
  }
  // gather z_q = E[idx]: wave copies 16 rows x 256 floats, 8 floats/lane/row
#pragma unroll 1
  for (int j = 0; j < 16; ++j) {
    int id = sIdx[wave][j];
    const v4f* src = (const v4f*)(E + (size_t)id * LATENT) + lane * 2;
    v4f* dst = (v4f*)(zq + (size_t)(m0 + j) * LATENT) + lane * 2;
    dst[0] = src[0];
    dst[1] = src[1];
  }
}

// ---------------------------------------------------------------------------
// Kernel 3: decoder x_recon = z_q @ Wd^T + bd, exact f32 WMMA. One wave per
// 16x256 strip (quarter of N=1024).
// ---------------------------------------------------------------------------
__global__ void __launch_bounds__(256) vq_decode(const float* __restrict__ zq,
                                                 const float* __restrict__ Wd,
                                                 const float* __restrict__ bd,
                                                 float* __restrict__ xr) {
  int lane = threadIdx.x & 31, wave = threadIdx.x >> 5;
  int w = blockIdx.x * 8 + wave;
  int m0 = (w >> 2) * 16;
  int n0 = (w & 3) * 256;
  int mrow = m0 + (lane & 15);
  int koff = 2 * (lane >> 4);
  v8f acc[16];
#pragma unroll
  for (int t = 0; t < 16; ++t) acc[t] = zero8();

  for (int k0 = 0; k0 < LATENT; k0 += 4) {
    v2f a = *(const v2f*)(zq + (size_t)mrow * LATENT + k0 + koff);
#pragma unroll
    for (int t = 0; t < 16; ++t) {
      int n = n0 + t * 16 + (lane & 15);
      v2f b = *(const v2f*)(Wd + (size_t)n * LATENT + k0 + koff);
      acc[t] = __builtin_amdgcn_wmma_f32_16x16x4_f32(false, a, false, b,
                                                     (short)0, acc[t], false, false);
    }
  }
#pragma unroll
  for (int t = 0; t < 16; ++t) {
    int n = n0 + t * 16 + (lane & 15);
    float bias = bd[n];
#pragma unroll
    for (int j = 0; j < 8; ++j) {
      int row = m0 + j + ((lane >> 4) << 3);
      xr[(size_t)row * INPUT_DIM + n] = acc[t][j] + bias;
    }
  }
}

extern "C" void kernel_launch(void* const* d_in, const int* in_sizes, int n_in,
                              void* d_out, int out_size, void* d_ws, size_t ws_size,
                              hipStream_t stream) {
  (void)in_sizes; (void)n_in; (void)out_size; (void)ws_size;
  const float* x  = (const float*)d_in[0];
  const float* We = (const float*)d_in[1];
  const float* be = (const float*)d_in[2];
  const float* E  = (const float*)d_in[3];
  const float* Wd = (const float*)d_in[4];
  const float* bd = (const float*)d_in[5];

  float* out = (float*)d_out;
  float* xr  = out;
  float* z   = out + (size_t)NTOT * INPUT_DIM;
  float* zq  = z + (size_t)NTOT * LATENT;
  int*   idx = (int*)(zq + (size_t)NTOT * LATENT);

  unsigned short* ehi = (unsigned short*)d_ws;                    // 4 MB
  unsigned short* elo = ehi + (size_t)NUMEMB * LATENT;            // 4 MB
  float* ehn = (float*)(elo + (size_t)NUMEMB * LATENT);           // 32 KB

  vq_prep<<<NUMEMB / 256, 256, 0, stream>>>(E, ehi, elo, ehn);
  vq_encode<<<(NTOT / 16) / 4, 128, 0, stream>>>(x, We, be, z);
  vq_assign<<<(NTOT / 16) / 8, 256, 0, stream>>>(z, ehi, elo, ehn, E, idx, zq);
  vq_decode<<<((NTOT / 16) * 4) / 8, 256, 0, stream>>>(zq, Wd, bd, xr);
}
